// TransformerEncoderCustomSA_54915451846834
// MI455X (gfx1250) — matrix operand order repaired and checked by reading, not compiled, not run on hardware
//
#include <hip/hip_runtime.h>

#define S_LEN   4096
#define C_DIM   128
#define HID_DIM 256
#define SCALE_ATT 0.25f
#define LN_EPS  1e-5f

typedef __attribute__((ext_vector_type(16))) __bf16 v16bf;
typedef __attribute__((ext_vector_type(8)))  __bf16 v8bf;
typedef __attribute__((ext_vector_type(8)))  float  v8f;
typedef __attribute__((ext_vector_type(4)))  float  v4f;

// ---------------- helpers ----------------

static __device__ inline float lane_xor_f(float v, int mask, int lane) {
    int idx = ((lane ^ mask) & 31) << 2;
    int r = __builtin_amdgcn_ds_bpermute(idx, __builtin_bit_cast(int, v));
    return __builtin_bit_cast(float, r);
}

static __device__ inline void wait_lds() {
    asm volatile("s_wait_dscnt 0x0" ::: "memory");
}

// A-matrix fragment (16x32 bf16, M x K), source row-major (K contiguous), ld in elements.
// Lane l (0-15): row M=l, K = k0+0..7 and k0+16..23 ; lane l+16: K = k0+8..15 and k0+24..31.
static __device__ inline v16bf load_fragA(const __bf16* base, int row0, int k0, int ld, int lane) {
    int l = lane & 15, hi = lane >> 4;
    const __bf16* p = base + (size_t)(row0 + l) * ld + k0 + hi * 8;
    v8bf lo = *(const v8bf*)(p);
    v8bf hh = *(const v8bf*)(p + 16);
    return __builtin_shufflevector(lo, hh, 0,1,2,3,4,5,6,7,8,9,10,11,12,13,14,15);
}

// B-matrix fragment (32x16 bf16, K x N), source stored N-major (column n's K run contiguous).
// Lane l (0-15): col N=l, K = k0+0..7 and k0+16..23 ; lane l+16: col N=l, K = k0+8..15 and k0+24..31.
static __device__ inline v16bf load_fragB(const __bf16* base, int col0, int k0, int ld, int lane) {
    int l = lane & 15, hi = lane >> 4;
    const __bf16* p = base + (size_t)(col0 + l) * ld + k0 + hi * 8;
    v8bf lo = *(const v8bf*)(p);
    v8bf hh = *(const v8bf*)(p + 16);
    return __builtin_shufflevector(lo, hh, 0,1,2,3,4,5,6,7,8,9,10,11,12,13,14,15);
}

static __device__ inline v8f wmma_bf16(v16bf a, v16bf b, v8f c) {
    return __builtin_amdgcn_wmma_f32_16x16x32_bf16(false, a, false, b, (short)0, c, false, false);
}

// first WMMA of a chain: literal-zero C so the backend can use inline SRC2=0
static __device__ inline v8f wmma_bf16_z(v16bf a, v16bf b) {
    v8f z = {0.f, 0.f, 0.f, 0.f, 0.f, 0.f, 0.f, 0.f};
    return __builtin_amdgcn_wmma_f32_16x16x32_bf16(false, a, false, b, (short)0, z, false, false);
}

static __device__ inline float gelu_exact(float x) {
    return 0.5f * x * (1.0f + erff(x * 0.70710678118654752f));
}

// ---------------- kernel 0: weight transpose + bf16 convert ----------------

__global__ __launch_bounds__(256) void k_prep(
    const float* __restrict__ wq, const float* __restrict__ wk,
    const float* __restrict__ wv, const float* __restrict__ wo,
    const float* __restrict__ w1, const float* __restrict__ w2,
    __bf16* __restrict__ wqT, __bf16* __restrict__ wkT,
    __bf16* __restrict__ wvT, __bf16* __restrict__ woT,
    __bf16* __restrict__ w1T, __bf16* __restrict__ w2T)
{
    int i = blockIdx.x * 256 + threadIdx.x;
    if (i < C_DIM * C_DIM) {          // [128][128] -> T
        int n = i >> 7, k = i & 127;
        wqT[i] = (__bf16)wq[k * C_DIM + n];
        wkT[i] = (__bf16)wk[k * C_DIM + n];
        wvT[i] = (__bf16)wv[k * C_DIM + n];
        woT[i] = (__bf16)wo[k * C_DIM + n];
    }
    if (i < HID_DIM * C_DIM) {
        { int n = i >> 7, k = i & 127; w1T[i] = (__bf16)w1[k * HID_DIM + n]; } // w1T[256][128]
        { int n = i >> 8, k = i & 255; w2T[i] = (__bf16)w2[k * C_DIM  + n]; } // w2T[128][256]
    }
}

// ---------------- kernel 1: LN1 + QKV projections ----------------

__global__ __launch_bounds__(128) void k_ln_qkv(
    const float* __restrict__ x, const float* __restrict__ ln1w, const float* __restrict__ ln1b,
    const __bf16* __restrict__ wqT, const float* __restrict__ bq,
    const __bf16* __restrict__ wkT, const float* __restrict__ bk,
    const __bf16* __restrict__ wvT, const float* __restrict__ bv,
    __bf16* __restrict__ Q, __bf16* __restrict__ K, __bf16* __restrict__ Vt)
{
    __shared__ __bf16 stage_s[4][16 * C_DIM];
    int lane = threadIdx.x & 31;
    int wid  = threadIdx.x >> 5;
    int tile = blockIdx.x * 4 + wid;
    int b    = tile >> 8;
    int srow = (tile & 255) << 4;
    int l15  = lane & 15, hi = lane >> 4;
    __bf16* st = stage_s[wid];
    const float* xb = x + (size_t)b * C_DIM * S_LEN;

    // ---- LayerNorm over C, staged as bf16 [16 x 128] ----
    float vals[4][16];
    #pragma unroll
    for (int j = 0; j < 4; ++j) {
        int c = lane + j * 32;
        const float* p = xb + (size_t)c * S_LEN + srow;
        #pragma unroll
        for (int r = 0; r < 16; ++r) vals[j][r] = p[r];
    }
    #pragma unroll
    for (int r = 0; r < 16; ++r) {
        float s = 0.f, s2 = 0.f;
        #pragma unroll
        for (int j = 0; j < 4; ++j) { float v = vals[j][r]; s += v; s2 += v * v; }
        #pragma unroll
        for (int m = 1; m < 32; m <<= 1) { s += lane_xor_f(s, m, lane); s2 += lane_xor_f(s2, m, lane); }
        float mu   = s * (1.f / 128.f);
        float var  = s2 * (1.f / 128.f) - mu * mu;
        float rstd = rsqrtf(var + LN_EPS);
        #pragma unroll
        for (int j = 0; j < 4; ++j) {
            int c = lane + j * 32;
            st[r * C_DIM + c] = (__bf16)((vals[j][r] - mu) * rstd * ln1w[c] + ln1b[c]);
        }
    }
    wait_lds();

    v16bf a[4];
    #pragma unroll
    for (int kc = 0; kc < 4; ++kc) a[kc] = load_fragA(st, 0, kc * 32, C_DIM, lane);

    const __bf16* wT[3] = { wqT, wkT, wvT };
    const float*  bb[3] = { bq, bk, bv };
    #pragma unroll
    for (int m = 0; m < 3; ++m) {
        #pragma unroll
        for (int n = 0; n < 8; ++n) {
            int cb = n * 16;
            float bias = bb[m][cb + l15];
            v8f acc = wmma_bf16_z(a[0], load_fragB(wT[m], cb, 0, C_DIM, lane));
            #pragma unroll
            for (int kc = 1; kc < 4; ++kc)
                acc = wmma_bf16(a[kc], load_fragB(wT[m], cb, kc * 32, C_DIM, lane), acc);
            if (m == 2) {  // V stored transposed: Vt[(b*C + c)*S + s]
                v8bf pk;
                #pragma unroll
                for (int v = 0; v < 8; ++v) pk[v] = (__bf16)(acc[v] + bias);
                *(v8bf*)(Vt + ((size_t)(b * C_DIM + cb + l15)) * S_LEN + srow + hi * 8) = pk;
            } else {       // Q/K row-major bf16 [B*S][128]
                __bf16* dst = (m == 0 ? Q : K) + (size_t)(b * S_LEN + srow) * C_DIM + cb + l15;
                #pragma unroll
                for (int v = 0; v < 8; ++v) dst[(size_t)(v + 8 * hi) * C_DIM] = (__bf16)(acc[v] + bias);
            }
        }
    }
}

// ---------------- kernel 2: per-key-column softmax stats (softmax over q axis) ----------------
// 4 waves/block handle 4 adjacent key blocks of the SAME batch; the streamed Q tiles are
// identical across waves, so they are staged cooperatively in LDS (double-buffered).

__global__ __launch_bounds__(128) void k_colstats(
    const __bf16* __restrict__ Q, const __bf16* __restrict__ K,
    float* __restrict__ mbuf, float* __restrict__ lbuf)
{
    __shared__ __bf16 qstage[2][16 * C_DIM];   // 2 x 4 KiB
    int tid  = threadIdx.x;
    int lane = tid & 31;
    int wid  = tid >> 5;
    int b     = blockIdx.x >> 6;               // 64 blocks per batch
    int kbase = (((blockIdx.x & 63) * 4 + wid)) << 4;
    const __bf16* Qb = Q + (size_t)b * S_LEN * C_DIM;
    const __bf16* Kb = K + (size_t)b * S_LEN * C_DIM;

    v16bf bk[4];
    #pragma unroll
    for (int kc = 0; kc < 4; ++kc) bk[kc] = load_fragB(Kb, kbase, kc * 32, C_DIM, lane);

    // preload Q tile 0 (16x128 bf16 = 4 KiB, contiguous)
    {
        const uint4* src = (const uint4*)Qb;
        uint4* dst = (uint4*)qstage[0];
        dst[tid]       = src[tid];
        dst[tid + 128] = src[tid + 128];
    }
    __syncthreads();

    float M = -1e30f, L = 0.f;
    for (int qb = 0; qb < 256; ++qb) {
        int cur = qb & 1;
        if (qb + 1 < 256) {                    // prefetch next tile into other buffer
            const uint4* src = (const uint4*)(Qb + (size_t)(qb + 1) * 16 * C_DIM);
            uint4* dst = (uint4*)qstage[cur ^ 1];
            dst[tid]       = src[tid];
            dst[tid + 128] = src[tid + 128];
        }
        const __bf16* qs = qstage[cur];
        v8f acc = wmma_bf16_z(load_fragA(qs, 0, 0, C_DIM, lane), bk[0]);
        #pragma unroll
        for (int kc = 1; kc < 4; ++kc)
            acc = wmma_bf16(load_fragA(qs, 0, kc * 32, C_DIM, lane), bk[kc], acc);

        float tmax = -1e30f;
        #pragma unroll
        for (int v = 0; v < 8; ++v) { float sv = acc[v] * SCALE_ATT; acc[v] = sv; tmax = fmaxf(tmax, sv); }
        tmax = fmaxf(tmax, lane_xor_f(tmax, 16, lane));   // combine rows M=v and M=v+8 halves
        float newM = fmaxf(M, tmax);
        float tsum = 0.f;
        #pragma unroll
        for (int v = 0; v < 8; ++v) tsum += __expf(acc[v] - newM);
        tsum += lane_xor_f(tsum, 16, lane);
        L = L * __expf(M - newM) + tsum;
        M = newM;
        __syncthreads();                       // reads of cur done; next iter may overwrite it
    }
    if (lane < 16) {
        mbuf[b * S_LEN + kbase + lane] = M;
        lbuf[b * S_LEN + kbase + lane] = L;
    }
}

// ---------------- kernel 3: attention output + wo + residual ----------------
// 4 waves/block handle 4 adjacent q tiles of the SAME batch and walk the key dimension
// together: K tiles (32x128, contiguous 8 KiB) and V^T tiles (128x32) are staged in LDS
// cooperatively, double-buffered, one workgroup barrier per iteration.

__global__ __launch_bounds__(128) void k_attn_out(
    const float* __restrict__ x,
    const __bf16* __restrict__ Q, const __bf16* __restrict__ K, const __bf16* __restrict__ Vt,
    const float* __restrict__ mbuf, const float* __restrict__ lbuf,
    const __bf16* __restrict__ woT, const float* __restrict__ bo,
    float* __restrict__ t2)
{
    __shared__ __bf16 kstage[2][32 * C_DIM];   // 2 x 8 KiB
    __shared__ __bf16 vstage[2][C_DIM * 32];   // 2 x 8 KiB
    __shared__ __bf16 pstage_s[4][16 * 32];
    __shared__ __bf16 ostage_s[4][16 * C_DIM];
    int tid  = threadIdx.x;
    int lane = tid & 31;
    int wid  = tid >> 5;
    int tile = blockIdx.x * 4 + wid;
    int b    = tile >> 8;
    int srow = (tile & 255) << 4;
    int l15  = lane & 15, hi = lane >> 4;
    __bf16* pst = pstage_s[wid];
    __bf16* ost = ostage_s[wid];
    const __bf16* Qb = Q  + (size_t)b * S_LEN * C_DIM;
    const __bf16* Kb = K  + (size_t)b * S_LEN * C_DIM;
    const __bf16* Vb = Vt + (size_t)b * C_DIM * S_LEN;

    v16bf aq[4];
    #pragma unroll
    for (int kc = 0; kc < 4; ++kc) aq[kc] = load_fragA(Qb, srow, kc * 32, C_DIM, lane);

    v8f acco[8];
    #pragma unroll
    for (int n = 0; n < 8; ++n)
        #pragma unroll
        for (int v = 0; v < 8; ++v) acco[n][v] = 0.f;

    // cooperative tile stager: K tile rows k0..k0+31 (contiguous), V^T rows c=0..127 (64B each)
    auto stage_tiles = [&](int k0, int bufi) {
        const uint4* ks = (const uint4*)(Kb + (size_t)k0 * C_DIM);
        uint4* kd = (uint4*)kstage[bufi];
        #pragma unroll
        for (int j = 0; j < 4; ++j) kd[tid + j * 128] = ks[tid + j * 128];
        const uint4* vs = (const uint4*)(Vb + (size_t)tid * S_LEN + k0);
        uint4* vd = (uint4*)(vstage[bufi] + tid * 32);
        #pragma unroll
        for (int j = 0; j < 4; ++j) vd[j] = vs[j];
    };

    stage_tiles(0, 0);
    __syncthreads();

    for (int it = 0; it < S_LEN / 32; ++it) {
        int k0  = it * 32;
        int cur = it & 1;
        if (it + 1 < S_LEN / 32) stage_tiles(k0 + 32, cur ^ 1);
        const __bf16* Ks = kstage[cur];
        const __bf16* Vs = vstage[cur];

        #pragma unroll
        for (int t = 0; t < 2; ++t) {
            v8f sc = wmma_bf16_z(aq[0], load_fragB(Ks, t * 16, 0, C_DIM, lane));
            #pragma unroll
            for (int kc = 1; kc < 4; ++kc)
                sc = wmma_bf16(aq[kc], load_fragB(Ks, t * 16, kc * 32, C_DIM, lane), sc);
            int col = k0 + t * 16 + l15;
            float mm = mbuf[b * S_LEN + col];
            float rz = 1.f / lbuf[b * S_LEN + col];
            #pragma unroll
            for (int v = 0; v < 8; ++v) {
                float p = __expf(sc[v] * SCALE_ATT - mm) * rz;    // column-normalized prob
                pst[(v + 8 * hi) * 32 + t * 16 + l15] = (__bf16)p;
            }
        }
        wait_lds();
        v16bf ap = load_fragA(pst, 0, 0, 32, lane);               // D-layout -> A-layout via LDS
        #pragma unroll
        for (int n = 0; n < 8; ++n)
            acco[n] = wmma_bf16(ap, load_fragB(Vs, n * 16, 0, 32, lane), acco[n]);
        __syncthreads();                                           // buffers free for next iter
    }

    // restage att_out, then x woT + bo + residual(x^T)
    #pragma unroll
    for (int n = 0; n < 8; ++n)
        #pragma unroll
        for (int v = 0; v < 8; ++v)
            ost[(v + 8 * hi) * C_DIM + n * 16 + l15] = (__bf16)acco[n][v];
    wait_lds();
    v16bf ao[4];
    #pragma unroll
    for (int kc = 0; kc < 4; ++kc) ao[kc] = load_fragA(ost, 0, kc * 32, C_DIM, lane);

    const float* xb = x + (size_t)b * C_DIM * S_LEN;
    #pragma unroll
    for (int n = 0; n < 8; ++n) {
        int c = n * 16 + l15;
        float bias = bo[c];
        const float* resp = xb + (size_t)c * S_LEN + srow + hi * 8;
        v8f acc;
        #pragma unroll
        for (int v = 0; v < 8; ++v) acc[v] = bias + resp[v];
        #pragma unroll
        for (int kc = 0; kc < 4; ++kc)
            acc = wmma_bf16(ao[kc], load_fragB(woT, n * 16, kc * 32, C_DIM, lane), acc);
        float* dst = t2 + (size_t)(b * S_LEN + srow) * C_DIM + c;
        #pragma unroll
        for (int v = 0; v < 8; ++v) dst[(size_t)(v + 8 * hi) * C_DIM] = acc[v];
    }
}

// ---------------- kernel 4: LN2 + MLP + GELUs + residual + transposed store ----------------

__global__ __launch_bounds__(128) void k_mlp(
    const float* __restrict__ t2,
    const float* __restrict__ ln2w, const float* __restrict__ ln2b,
    const __bf16* __restrict__ w1T, const float* __restrict__ b1,
    const __bf16* __restrict__ w2T, const float* __restrict__ b2,
    float* __restrict__ out)
{
    __shared__ __bf16 xstage_s[4][16 * C_DIM];
    __shared__ __bf16 hstage_s[4][16 * HID_DIM];
    int lane = threadIdx.x & 31;
    int wid  = threadIdx.x >> 5;
    int tile = blockIdx.x * 4 + wid;
    int b    = tile >> 8;
    int srow = (tile & 255) << 4;
    int l15  = lane & 15, hi = lane >> 4;
    __bf16* xst = xstage_s[wid];
    __bf16* hst = hstage_s[wid];
    const float* trow0 = t2 + (size_t)(b * S_LEN + srow) * C_DIM;

    // ---- LN2 (t2 row-major, coalesced) ----
    #pragma unroll
    for (int r = 0; r < 16; ++r) {
        const float* p = trow0 + (size_t)r * C_DIM;
        float lv[4]; float s = 0.f, s2 = 0.f;
        #pragma unroll
        for (int j = 0; j < 4; ++j) { lv[j] = p[lane * 4 + j]; s += lv[j]; s2 += lv[j] * lv[j]; }
        #pragma unroll
        for (int m = 1; m < 32; m <<= 1) { s += lane_xor_f(s, m, lane); s2 += lane_xor_f(s2, m, lane); }
        float mu   = s * (1.f / 128.f);
        float var  = s2 * (1.f / 128.f) - mu * mu;
        float rstd = rsqrtf(var + LN_EPS);
        #pragma unroll
        for (int j = 0; j < 4; ++j) {
            int c = lane * 4 + j;
            xst[r * C_DIM + c] = (__bf16)((lv[j] - mu) * rstd * ln2w[c] + ln2b[c]);
        }
    }
    wait_lds();

    v16bf ax[4];
    #pragma unroll
    for (int kc = 0; kc < 4; ++kc) ax[kc] = load_fragA(xst, 0, kc * 32, C_DIM, lane);

    // ---- hidden = gelu(xl @ w1 + b1), staged bf16 [16 x 256] ----
    #pragma unroll
    for (int h = 0; h < 2; ++h) {
        #pragma unroll
        for (int n = 0; n < 8; ++n) {
            int cb = h * 128 + n * 16;
            float bias = b1[cb + l15];
            v8f acc = wmma_bf16_z(ax[0], load_fragB(w1T, cb, 0, C_DIM, lane));
            #pragma unroll
            for (int kc = 1; kc < 4; ++kc)
                acc = wmma_bf16(ax[kc], load_fragB(w1T, cb, kc * 32, C_DIM, lane), acc);
            #pragma unroll
            for (int v = 0; v < 8; ++v)
                hst[(v + 8 * hi) * HID_DIM + cb + l15] = (__bf16)gelu_exact(acc[v] + bias);
        }
    }
    wait_lds();

    v16bf ah[8];
    #pragma unroll
    for (int kc = 0; kc < 8; ++kc) ah[kc] = load_fragA(hst, 0, kc * 32, HID_DIM, lane);

    // ---- y = gelu(h @ w2 + b2) + t2, transposed store to [B][C][S] ----
    #pragma unroll
    for (int n = 0; n < 8; ++n) {
        int c = n * 16 + l15;
        float bias = b2[c];
        v8f acc = wmma_bf16_z(ah[0], load_fragB(w2T, n * 16, 0, HID_DIM, lane));
        #pragma unroll
        for (int kc = 1; kc < 8; ++kc)
            acc = wmma_bf16(ah[kc], load_fragB(w2T, n * 16, kc * 32, HID_DIM, lane), acc);
        float o[8];
        const float* resp = trow0 + c;
        #pragma unroll
        for (int v = 0; v < 8; ++v)
            o[v] = gelu_exact(acc[v] + bias) + resp[(size_t)(v + 8 * hi) * C_DIM];
        float* dst = out + ((size_t)(b * C_DIM + c)) * S_LEN + srow + hi * 8;
        *(v4f*)(dst)     = (v4f){o[0], o[1], o[2], o[3]};
        *(v4f*)(dst + 4) = (v4f){o[4], o[5], o[6], o[7]};
    }
}

// ---------------- launch ----------------

extern "C" void kernel_launch(void* const* d_in, const int* in_sizes, int n_in,
                              void* d_out, int out_size, void* d_ws, size_t ws_size,
                              hipStream_t stream) {
    (void)in_sizes; (void)n_in; (void)out_size; (void)ws_size;
    const float* x    = (const float*)d_in[0];
    const float* ln1w = (const float*)d_in[1];
    const float* ln1b = (const float*)d_in[2];
    const float* wq   = (const float*)d_in[3];
    const float* bq   = (const float*)d_in[4];
    const float* wk   = (const float*)d_in[5];
    const float* bk   = (const float*)d_in[6];
    const float* wv   = (const float*)d_in[7];
    const float* bv   = (const float*)d_in[8];
    const float* wo   = (const float*)d_in[9];
    const float* bo   = (const float*)d_in[10];
    const float* ln2w = (const float*)d_in[11];
    const float* ln2b = (const float*)d_in[12];
    const float* w1   = (const float*)d_in[13];
    const float* b1   = (const float*)d_in[14];
    const float* w2   = (const float*)d_in[15];
    const float* b2   = (const float*)d_in[16];
    float* out = (float*)d_out;

    char* ws = (char*)d_ws;
    const size_t MB = (size_t)1 << 20;
    __bf16* Q   = (__bf16*)(ws);                              //  4 MiB  [B*S][128] bf16
    __bf16* Km  = (__bf16*)(ws + 4  * MB);                    //  4 MiB
    __bf16* Vt  = (__bf16*)(ws + 8  * MB);                    //  4 MiB  [B][128][S] bf16
    float*  mb  = (float*) (ws + 12 * MB);                    // 64 KiB
    float*  lb  = (float*) (ws + 12 * MB + 65536);            // 64 KiB
    float*  t2  = (float*) (ws + 13 * MB);                    //  8 MiB  [B*S][128] f32
    __bf16* wqT = (__bf16*)(ws + 21 * MB);
    __bf16* wkT = wqT + 128 * 128;
    __bf16* wvT = wkT + 128 * 128;
    __bf16* woT = wvT + 128 * 128;
    __bf16* w1T = woT + 128 * 128;                            // [256][128]
    __bf16* w2T = w1T + 256 * 128;                            // [128][256]

    k_prep    <<<128, 256, 0, stream>>>(wq, wk, wv, wo, w1, w2, wqT, wkT, wvT, woT, w1T, w2T);
    k_ln_qkv  <<<256, 128, 0, stream>>>(x, ln1w, ln1b, wqT, bq, wkT, bk, wvT, bv, Q, Km, Vt);
    k_colstats<<<256, 128, 0, stream>>>(Q, Km, mb, lb);
    k_attn_out<<<256, 128, 0, stream>>>(x, Q, Km, Vt, mb, lb, woT, bo, t2);
    k_mlp     <<<256, 128, 0, stream>>>(t2, ln2w, ln2b, w1T, b1, w2T, b2, out);
}
